// RetentiveSelfAttention_40286793236541
// MI455X (gfx1250) — compile-verified, matrix operand order checked by reading, not run on hardware
//
#include <hip/hip_runtime.h>

// ---------------------------------------------------------------------------
// Types / helpers
// ---------------------------------------------------------------------------
typedef __attribute__((ext_vector_type(16))) __bf16 v16bf;
typedef __attribute__((ext_vector_type(8)))  float  v8f;
typedef __attribute__((ext_vector_type(4)))  unsigned int u32x4;
typedef __attribute__((ext_vector_type(4)))  float  f32x4;

union Frag16 { v16bf bf; u32x4 q[2]; unsigned short h[16]; };

constexpr int NSEQ  = 48 * 48;       // 2304
constexpr int HEADS = 12;
constexpr int DHEAD = 64;
constexpr int DIM   = 768;
constexpr int BATCH = 4;
constexpr int MROWS = BATCH * NSEQ;  // 9216

static __device__ __forceinline__ u32x4 ld16(const unsigned short* p) {
  return *(const u32x4*)p;
}
static __device__ __forceinline__ unsigned short f2bf(float f) {
  unsigned int u = __float_as_uint(f);
  u += 0x7FFFu + ((u >> 16) & 1u);           // round-to-nearest-even
  return (unsigned short)(u >> 16);
}
static __device__ __forceinline__ v8f wmma_bf16(const Frag16& a, const Frag16& b, v8f c) {
  return __builtin_amdgcn_wmma_f32_16x16x32_bf16(false, a.bf, false, b.bf,
                                                 (short)0, c, false, false);
}

// ---------------------------------------------------------------------------
// fp32 -> bf16 convert (optionally scaled; Wk gets SCALING*log2e folded in)
// ---------------------------------------------------------------------------
__global__ void cvt_bf16(const float* __restrict__ s, unsigned short* __restrict__ d,
                         int n, float scale) {
  int i = blockIdx.x * blockDim.x + threadIdx.x;
  if (i < n) d[i] = f2bf(s[i] * scale);
}

// ---------------------------------------------------------------------------
// Tiled WMMA GEMM:  C[m,c] = sum_k A[m,k] * W[c,k] + bias[c]*biasScale
// A: MxK bf16 row-major, W: NxK bf16 row-major (nn.Linear weight layout).
// One wave owns a 32x64 output tile (2x4 WMMA tiles).
// mode 0: store bf16 at [b,h,n,64]   (Q,K)
// mode 1: store bf16 at [b,h,64,n]   (V transposed)
// mode 2: store f32 row-major MxN    (final output)
// ---------------------------------------------------------------------------
__global__ __launch_bounds__(128) void gemm_wmma(
    const unsigned short* __restrict__ A, const unsigned short* __restrict__ W,
    const float* __restrict__ bias, float biasScale,
    void* __restrict__ dst, int M, int K, int N, int mode) {
  const int wave = threadIdx.x >> 5;
  const int lane = threadIdx.x & 31;
  const int l15  = lane & 15;
  const int off8 = (lane & 16) ? 8 : 0;      // k-chunk offset per A/B layout

  const int nT = N / 64;
  const int mT = M / 32;
  const int tile = blockIdx.x * 4 + wave;
  const int mIdx = tile / nT;
  const int nIdx = tile % nT;
  if (mIdx >= mT) return;                    // wave-uniform: EXEC stays all-ones for WMMA

  const long mBase = (long)mIdx * 32;
  const long nBase = (long)nIdx * 64;

  v8f acc[2][4];
#pragma unroll
  for (int r = 0; r < 2; ++r)
#pragma unroll
    for (int c = 0; c < 4; ++c) acc[r][c] = (v8f)0.0f;

  for (int kk = 0; kk < K; kk += 32) {
    Frag16 af[2], bf[4];
#pragma unroll
    for (int rt = 0; rt < 2; ++rt) {
      const unsigned short* ap = A + (size_t)(mBase + rt * 16 + l15) * K + kk + off8;
      af[rt].q[0] = ld16(ap);
      af[rt].q[1] = ld16(ap + 16);
    }
#pragma unroll
    for (int ct = 0; ct < 4; ++ct) {
      const unsigned short* wp = W + (size_t)(nBase + ct * 16 + l15) * K + kk + off8;
      bf[ct].q[0] = ld16(wp);
      bf[ct].q[1] = ld16(wp + 16);
    }
#pragma unroll
    for (int rt = 0; rt < 2; ++rt)
#pragma unroll
      for (int ct = 0; ct < 4; ++ct)
        acc[rt][ct] = wmma_bf16(af[rt], bf[ct], acc[rt][ct]);
  }

  // epilogue: C/D layout: VGPR slot r -> row mBase+rt*16+r+off8, col = nBase+ct*16+l15
#pragma unroll
  for (int rt = 0; rt < 2; ++rt) {
#pragma unroll
    for (int ct = 0; ct < 4; ++ct) {
      const long col = nBase + ct * 16 + l15;
      const float bv = bias[col] * biasScale;
#pragma unroll
      for (int r = 0; r < 8; ++r) {
        const long row = mBase + rt * 16 + r + off8;
        const float v = acc[rt][ct][r] + bv;
        if (mode == 2) {
          ((float*)dst)[(size_t)row * N + col] = v;
        } else {
          const long b_ = row / NSEQ, n_ = row % NSEQ;
          const long h_ = col >> 6, d_ = col & 63;
          size_t idx;
          if (mode == 0) idx = (((size_t)(b_ * HEADS + h_) * NSEQ) + n_) * DHEAD + d_;
          else           idx = (((size_t)(b_ * HEADS + h_) * DHEAD) + d_) * NSEQ + n_;
          ((unsigned short*)dst)[idx] = f2bf(v);
        }
      }
    }
  }
}

// ---------------------------------------------------------------------------
// Fused flash-style retentive attention.
// Block = 4 waves, wave w = batch w; block handles (head h, 16 query rows).
// Sᵀ = K_j · Q_iᵀ so the score D-layout matches the P A-operand layout for PV;
// decay (post-softmax mask) rides inside the PV accumulation, the softmax
// denominator uses undamped p. Scores already include SCALING*log2(e) (folded
// into Wk), so exp2 is exact softmax.
// ---------------------------------------------------------------------------
__global__ __launch_bounds__(128) void attn_kernel(
    const unsigned short* __restrict__ Q,   // [b,h,n,64] bf16
    const unsigned short* __restrict__ K,   // [b,h,n,64] bf16
    const unsigned short* __restrict__ Vt,  // [b,h,64,n] bf16
    const float* __restrict__ decay,        // [h,n,n] f32
    unsigned short* __restrict__ att) {     // [b,n,768] bf16
  const int b    = threadIdx.x >> 5;
  const int lane = threadIdx.x & 31;
  const int l15  = lane & 15;
  const int off8 = (lane & 16) ? 8 : 0;
  const int h     = blockIdx.x / (NSEQ / 16);
  const int iBase = (blockIdx.x % (NSEQ / 16)) * 16;

  // Q B-operand fragments: constant across the whole j loop.
  const unsigned short* qrow = Q + ((size_t)(b * HEADS + h) * NSEQ + iBase + l15) * DHEAD;
  Frag16 qf0, qf1;
  qf0.q[0] = ld16(qrow + off8);       qf0.q[1] = ld16(qrow + off8 + 16);
  qf1.q[0] = ld16(qrow + off8 + 32);  qf1.q[1] = ld16(qrow + off8 + 48);

  const unsigned short* kbase = K  + (size_t)(b * HEADS + h) * NSEQ * DHEAD;
  const unsigned short* vbase = Vt + (size_t)(b * HEADS + h) * DHEAD * NSEQ;
  const float* dbase = decay + ((size_t)h * NSEQ + iBase + l15) * NSEQ + off8;

  float m = -3.0e38f;   // running max (log2 domain), per lane == per query row
  float l = 0.0f;       // running softmax denominator
  v8f acc[4];
#pragma unroll
  for (int t = 0; t < 4; ++t) acc[t] = (v8f)0.0f;

  for (int j0 = 0; j0 < NSEQ; j0 += 32) {
    // ---- Sᵀ tiles for j-blocks j0 and j0+16 (each = 2 WMMAs over d=64)
    Frag16 kf;
    v8f s0 = (v8f)0.0f, s1 = (v8f)0.0f;
    const unsigned short* kr0 = kbase + (size_t)(j0 + l15) * DHEAD + off8;
    kf.q[0] = ld16(kr0);      kf.q[1] = ld16(kr0 + 16);
    s0 = wmma_bf16(kf, qf0, s0);
    kf.q[0] = ld16(kr0 + 32); kf.q[1] = ld16(kr0 + 48);
    s0 = wmma_bf16(kf, qf1, s0);
    const unsigned short* kr1 = kbase + (size_t)(j0 + 16 + l15) * DHEAD + off8;
    kf.q[0] = ld16(kr1);      kf.q[1] = ld16(kr1 + 16);
    s1 = wmma_bf16(kf, qf0, s1);
    kf.q[0] = ld16(kr1 + 32); kf.q[1] = ld16(kr1 + 48);
    s1 = wmma_bf16(kf, qf1, s1);

    if (j0 + 32 < NSEQ) {   // prefetch next iteration's K rows
      __builtin_prefetch(kbase + (size_t)(j0 + 32 + l15) * DHEAD, 0, 3);
      __builtin_prefetch(kbase + (size_t)(j0 + 48 + l15) * DHEAD, 0, 3);
    }

    // ---- decay tiles (f32, contiguous 8 per lane per j-block)
    f32x4 d0a = *(const f32x4*)(dbase + j0);
    f32x4 d0b = *(const f32x4*)(dbase + j0 + 4);
    f32x4 d1a = *(const f32x4*)(dbase + j0 + 16);
    f32x4 d1b = *(const f32x4*)(dbase + j0 + 20);

    // ---- V B-operand fragments (Vt rows contiguous along j)
    Frag16 vf[4];
#pragma unroll
    for (int t = 0; t < 4; ++t) {
      const unsigned short* vr = vbase + (size_t)(t * 16 + l15) * NSEQ + j0 + off8;
      vf[t].q[0] = ld16(vr);
      vf[t].q[1] = ld16(vr + 16);
    }

    // ---- online softmax (lane pair n / n^16 shares one query row)
    float lm = s0[0];
#pragma unroll
    for (int r = 1; r < 8; ++r) lm = fmaxf(lm, s0[r]);
#pragma unroll
    for (int r = 0; r < 8; ++r) lm = fmaxf(lm, s1[r]);
    lm = fmaxf(lm, __shfl_xor(lm, 16, 32));
    const float nm = fmaxf(m, lm);
    const float sc = __builtin_amdgcn_exp2f(m - nm);

    float p0[8], p1[8], ps = 0.0f;
#pragma unroll
    for (int r = 0; r < 8; ++r) {
      p0[r] = __builtin_amdgcn_exp2f(s0[r] - nm);
      p1[r] = __builtin_amdgcn_exp2f(s1[r] - nm);
      ps += p0[r] + p1[r];
    }
    ps += __shfl_xor(ps, 16, 32);
    l = l * sc + ps;
    m = nm;

    // ---- pack p*decay straight into the PV A-operand layout (no transpose!)
    Frag16 pf;
#pragma unroll
    for (int r = 0; r < 4; ++r) {
      pf.h[r]      = f2bf(p0[r]     * d0a[r]);
      pf.h[4 + r]  = f2bf(p0[4 + r] * d0b[r]);
      pf.h[8 + r]  = f2bf(p1[r]     * d1a[r]);
      pf.h[12 + r] = f2bf(p1[4 + r] * d1b[r]);
    }

    // ---- rescale accumulator rows (row of slot r lives in lane r+off8)
#pragma unroll
    for (int r = 0; r < 8; ++r) {
      const float scr = __shfl(sc, r + off8, 32);
#pragma unroll
      for (int t = 0; t < 4; ++t) acc[t][r] *= scr;
    }

    // ---- PV: acc[i, d-tile] += P(16x32) x V(32x16)
#pragma unroll
    for (int t = 0; t < 4; ++t) acc[t] = wmma_bf16(pf, vf[t], acc[t]);
  }

  // ---- normalize by softmax denominator and store att[b,n, h*64+d] (bf16)
  const float rinv = 1.0f / l;
#pragma unroll
  for (int r = 0; r < 8; ++r) {
    const float rn = __shfl(rinv, r + off8, 32);
    const size_t row = (size_t)b * NSEQ + iBase + r + off8;
#pragma unroll
    for (int t = 0; t < 4; ++t) {
      att[row * DIM + h * 64 + t * 16 + l15] = f2bf(acc[t][r] * rn);
    }
  }
}

// ---------------------------------------------------------------------------
// Host-side orchestration
// ---------------------------------------------------------------------------
extern "C" void kernel_launch(void* const* d_in, const int* in_sizes, int n_in,
                              void* d_out, int out_size, void* d_ws, size_t ws_size,
                              hipStream_t stream) {
  const float* x     = (const float*)d_in[0];
  const float* decay = (const float*)d_in[1];
  const float* Wq    = (const float*)d_in[2];
  const float* bq    = (const float*)d_in[3];
  const float* Wk    = (const float*)d_in[4];
  const float* bk    = (const float*)d_in[5];
  const float* Wv    = (const float*)d_in[6];
  const float* bv    = (const float*)d_in[7];
  const float* Wo    = (const float*)d_in[8];
  const float* bo    = (const float*)d_in[9];

  char* ws = (char*)d_ws;
  const size_t xbfB = (size_t)MROWS * DIM * 2;          // 14,155,776
  const size_t wB   = (size_t)DIM * DIM * 2;            //  1,179,648
  const size_t qkvB = (size_t)BATCH * HEADS * NSEQ * DHEAD * 2;  // 14,155,776
  unsigned short* xbf  = (unsigned short*)(ws);
  unsigned short* wqbf = (unsigned short*)(ws + xbfB);
  unsigned short* wkbf = (unsigned short*)(ws + xbfB + wB);
  unsigned short* wvbf = (unsigned short*)(ws + xbfB + 2 * wB);
  unsigned short* wobf = (unsigned short*)(ws + xbfB + 3 * wB);
  unsigned short* qbf  = (unsigned short*)(ws + xbfB + 4 * wB);
  unsigned short* kbf  = (unsigned short*)(ws + xbfB + 4 * wB + qkvB);
  unsigned short* vtbf = (unsigned short*)(ws + xbfB + 4 * wB + 2 * qkvB);
  unsigned short* attb = (unsigned short*)(ws + xbfB + 4 * wB + 3 * qkvB);

  // fold SCALING (=64^-0.5) and log2(e) into K so softmax uses raw exp2
  const float kscale = 0.125f * 1.44269504088896340736f;

  const int nx = MROWS * DIM;
  const int nw = DIM * DIM;
  cvt_bf16<<<(nx + 255) / 256, 256, 0, stream>>>(x,  xbf,  nx, 1.0f);
  cvt_bf16<<<(nw + 255) / 256, 256, 0, stream>>>(Wq, wqbf, nw, 1.0f);
  cvt_bf16<<<(nw + 255) / 256, 256, 0, stream>>>(Wk, wkbf, nw, kscale);
  cvt_bf16<<<(nw + 255) / 256, 256, 0, stream>>>(Wv, wvbf, nw, 1.0f);
  cvt_bf16<<<(nw + 255) / 256, 256, 0, stream>>>(Wo, wobf, nw, 1.0f);

  // QKV projections: 9216x768 x 768x768, tiles = (9216/32)*(768/64) = 3456 waves
  const int gemmBlocks = (MROWS / 32) * (DIM / 64) / 4;  // 864
  gemm_wmma<<<gemmBlocks, 128, 0, stream>>>(xbf, wqbf, bq, 1.0f,  qbf,  MROWS, DIM, DIM, 0);
  gemm_wmma<<<gemmBlocks, 128, 0, stream>>>(xbf, wkbf, bk, kscale, kbf, MROWS, DIM, DIM, 0);
  gemm_wmma<<<gemmBlocks, 128, 0, stream>>>(xbf, wvbf, bv, 1.0f,  vtbf, MROWS, DIM, DIM, 1);

  // fused attention: block = (head, 16 query rows), 4 waves = 4 batches
  attn_kernel<<<HEADS * (NSEQ / 16), 128, 0, stream>>>(qbf, kbf, vtbf, decay, attb);

  // output projection, f32 result straight to d_out
  gemm_wmma<<<gemmBlocks, 128, 0, stream>>>(attb, wobf, bo, 1.0f, d_out, MROWS, DIM, DIM, 2);
}
